// TransformerDecoder_56057913147407
// MI455X (gfx1250) — matrix operand order repaired and checked by reading, not compiled, not run on hardware
//
#include <hip/hip_runtime.h>

typedef __attribute__((ext_vector_type(16))) _Float16 v16h;
typedef __attribute__((ext_vector_type(8)))  _Float16 v8h;
typedef __attribute__((ext_vector_type(8)))  float    v8f;

constexpr int Bc  = 4;
constexpr int Sc  = 1024;
constexpr int Ec  = 1024;
constexpr int Hc  = 16;
constexpr int Ac  = 64;
constexpr int SKVc = 1024;
constexpr int Mr  = Bc * Sc;     // 4096 rows of [*, E]
constexpr int FFc = 4 * Ec;      // 4096

__device__ __forceinline__ v16h cat8(v8h lo, v8h hi) {
    v16h r;
#pragma unroll
    for (int i = 0; i < 8; ++i) { r[i] = lo[i]; r[i + 8] = hi[i]; }
    return r;
}

// ---------------------------------------------------------------------------
// Elementwise helpers
// ---------------------------------------------------------------------------
__global__ void add_cast_kernel(const float* __restrict__ a, const float* __restrict__ b,
                                float* __restrict__ of, _Float16* __restrict__ oh, int n) {
    int i = blockIdx.x * blockDim.x + threadIdx.x;
    if (i < n) { float v = a[i] + b[i]; of[i] = v; oh[i] = (_Float16)v; }
}

__global__ void cast_kernel(const float* __restrict__ a, _Float16* __restrict__ o, int n) {
    int i = blockIdx.x * blockDim.x + threadIdx.x;
    if (i < n) o[i] = (_Float16)a[i];
}

// W [H,E,A] f32  ->  out [E, H*A] f16   (out col = h*A + a)
__global__ void repack_qkv_kernel(const float* __restrict__ w, _Float16* __restrict__ o, int n) {
    int i = blockIdx.x * blockDim.x + threadIdx.x;
    if (i < n) {
        int col = i & (Ec - 1);
        int e   = i >> 10;
        int h   = col >> 6;
        int a   = col & 63;
        o[i] = (_Float16)w[((long)h * Ec + e) * Ac + a];
    }
}

// ---------------------------------------------------------------------------
// Generic tiled WMMA GEMM: C[M,N] = A[M,K] * B[K,N] (+bias) (+relu)
// f16 inputs, f32 accumulation, output f16 or f32.
// Block: 256 threads (8 waves). Tile: 128(M) x 128(N), K-step 32.
// Waves arranged 4(M) x 2(N); each wave owns a 32x64 patch = 2x4 WMMA tiles,
// so fragment traffic is 12 ds_load_b128 per 8 WMMAs.
// ---------------------------------------------------------------------------
template <int OUTF16, int RELU, int BIAS>
__global__ __launch_bounds__(256) void gemm_wmma(
    const _Float16* __restrict__ A, const _Float16* __restrict__ Bw,
    const float* __restrict__ bias, void* __restrict__ Cout,
    int M, int N, int K) {
    __shared__ alignas(16) _Float16 As[128][40];   // 80B stride, 16B aligned
    __shared__ alignas(16) _Float16 Bt[128][40];   // B transposed: Bt[n][k]

    const int t    = threadIdx.x;
    const int lane = t & 31;
    const int wave = t >> 5;        // 0..7
    const int wm   = wave >> 1;     // 0..3  (M direction, 32 rows each)
    const int wn   = wave & 1;      // 0..1  (N direction, 64 cols each)
    const int m0   = blockIdx.y * 128;
    const int n0   = blockIdx.x * 128;

    const int rq   = lane & 15;
    const int klo  = (lane < 16) ? 0 : 8;
    const int kb16 = (lane < 16) ? 0 : 16;
    const int hi8  = (lane < 16) ? 0 : 8;

    const int arow = t >> 1, acol = (t & 1) * 16;    // A loader: 32 halves / 2 threads
    const int brow = t >> 3, bcol = (t & 7) * 16;    // B loader: 16 halves / thread

    v8f acc[2][4];
#pragma unroll
    for (int i = 0; i < 2; ++i)
#pragma unroll
        for (int j = 0; j < 4; ++j) acc[i][j] = v8f{};

    for (int k0 = 0; k0 < K; k0 += 32) {
        const _Float16* asrc = A + (long)(m0 + arow) * K + k0 + acol;
        *(uint4*)&As[arow][acol]     = *(const uint4*)(asrc);
        *(uint4*)&As[arow][acol + 8] = *(const uint4*)(asrc + 8);

        const _Float16* bsrc = Bw + (long)(k0 + brow) * N + n0 + bcol;
        _Float16 tmp[16];
        *(uint4*)&tmp[0] = *(const uint4*)(bsrc);
        *(uint4*)&tmp[8] = *(const uint4*)(bsrc + 8);
#pragma unroll
        for (int j = 0; j < 16; ++j) Bt[bcol + j][brow] = tmp[j];
        __syncthreads();

        v16h af[2], bf[4];
#pragma unroll
        for (int mi = 0; mi < 2; ++mi) {
            int r = wm * 32 + mi * 16 + rq;
            v8h lo = *(const v8h*)&As[r][klo];
            v8h hi = *(const v8h*)&As[r][klo + 16];
            af[mi] = cat8(lo, hi);
        }
#pragma unroll
        for (int ni = 0; ni < 4; ++ni) {
            int c = wn * 64 + ni * 16 + rq;
            v8h lo = *(const v8h*)&Bt[c][klo];
            v8h hi = *(const v8h*)&Bt[c][klo + 16];
            bf[ni] = cat8(lo, hi);
        }
#pragma unroll
        for (int mi = 0; mi < 2; ++mi)
#pragma unroll
            for (int ni = 0; ni < 4; ++ni)
                acc[mi][ni] = __builtin_amdgcn_wmma_f32_16x16x32_f16(
                    false, af[mi], false, bf[ni], (short)0, acc[mi][ni], false, false);
        __syncthreads();
    }

#pragma unroll
    for (int mi = 0; mi < 2; ++mi)
#pragma unroll
        for (int ni = 0; ni < 4; ++ni)
#pragma unroll
            for (int v = 0; v < 8; ++v) {
                int r = m0 + wm * 32 + mi * 16 + v + hi8;
                int c = n0 + wn * 64 + ni * 16 + (lane & 15);
                float val = acc[mi][ni][v];
                if (BIAS) val += bias[c];
                if (RELU) val = fmaxf(val, 0.0f);
                if (OUTF16) ((_Float16*)Cout)[(long)r * N + c] = (_Float16)val;
                else        ((float*)Cout)[(long)r * N + c]    = val;
            }
}

// ---------------------------------------------------------------------------
// Flash attention: one wave handles a 16-query tile for one (h,b).
// Q: [4096][E] f16 (col = h*64 + a).  K/V: self layout (stride E) or
// cross layout (packed per (h,b), stride 64).  Output: col h*64+a of [4096][E].
// ---------------------------------------------------------------------------
__global__ __launch_bounds__(32) void flash_attn(
    const _Float16* __restrict__ Q,
    const _Float16* __restrict__ Kp,
    const _Float16* __restrict__ Vp,
    _Float16* __restrict__ outH,   // nullable (f16 out)
    float* __restrict__ outF,      // nullable (f32 out)
    int skv, int causal, int kvself, float scale) {
    const int lane = threadIdx.x;
    const int tile = blockIdx.x;
    const int hb   = blockIdx.y;
    const int b    = hb % Bc;
    const int h    = hb / Bc;
    const int q0   = tile * 16;

    const _Float16* Qb = Q + (long)(b * Sc) * Ec + h * Ac;
    const _Float16* Kb;
    const _Float16* Vb;
    long kvstride;
    if (kvself) {
        kvstride = Ec;
        Kb = Kp + (long)(b * Sc) * Ec + h * Ac;
        Vb = Vp + (long)(b * Sc) * Ec + h * Ac;
    } else {
        kvstride = Ac;
        long base = ((long)h * Bc + b) * SKVc * Ac;
        Kb = Kp + base;
        Vb = Vp + base;
    }

    __shared__ alignas(16) _Float16 Qs[16][80];
    __shared__ alignas(16) _Float16 Ks[32][80];
    __shared__ alignas(16) _Float16 Vt[64][40];   // transposed: Vt[a][k]
    __shared__ alignas(16) _Float16 Ps[16][40];

    // stage Q tile (16 x 64 halves)
    {
        int r  = lane & 15;
        int c0 = (lane < 16) ? 0 : 32;
        const _Float16* src = Qb + (long)(q0 + r) * Ec + c0;
#pragma unroll
        for (int i = 0; i < 4; ++i)
            *(uint4*)&Qs[r][c0 + i * 8] = *(const uint4*)(src + i * 8);
    }
    __syncthreads();

    const int rq   = lane & 15;
    const int klo  = (lane < 16) ? 0 : 8;
    const int kb16 = (lane < 16) ? 0 : 16;
    const int hi8  = (lane < 16) ? 0 : 8;
    const int colc = lane & 15;

    v16h aQ[2];
#pragma unroll
    for (int f = 0; f < 2; ++f) {
        v8h lo = *(const v8h*)&Qs[rq][f * 32 + klo];
        v8h hi = *(const v8h*)&Qs[rq][f * 32 + klo + 16];
        aQ[f] = cat8(lo, hi);
    }

    v8f acc[4];
#pragma unroll
    for (int n = 0; n < 4; ++n) acc[n] = v8f{};
    float m[8], l[8];
#pragma unroll
    for (int v = 0; v < 8; ++v) { m[v] = -3.0e38f; l[v] = 0.0f; }

    const int nkeys = causal ? (q0 + 16) : skv;
    const int nch   = (nkeys + 31) / 32;

    for (int ch = 0; ch < nch; ++ch) {
        const int k0 = ch * 32;
        // stage K chunk (32 keys x 64) + V chunk transposed
        {
            const _Float16* ks = Kb + (long)(k0 + lane) * kvstride;
#pragma unroll
            for (int i = 0; i < 8; ++i)
                *(uint4*)&Ks[lane][i * 8] = *(const uint4*)(ks + i * 8);
            const _Float16* vs = Vb + (long)(k0 + lane) * kvstride;
#pragma unroll
            for (int a = 0; a < 64; ++a) Vt[a][lane] = vs[a];
        }
        __syncthreads();

        // scores: two 16x16 tiles, contraction over A=64 in two K=32 WMMAs
        v8f s[2];
#pragma unroll
        for (int nt = 0; nt < 2; ++nt) {
            v8f c = v8f{};
#pragma unroll
            for (int f = 0; f < 2; ++f) {
                v8h lo = *(const v8h*)&Ks[nt * 16 + colc][f * 32 + klo];
                v8h hi = *(const v8h*)&Ks[nt * 16 + colc][f * 32 + klo + 16];
                v16h bK = cat8(lo, hi);
                c = __builtin_amdgcn_wmma_f32_16x16x32_f16(
                    false, aQ[f], false, bK, (short)0, c, false, false);
            }
            s[nt] = c;
        }

        // scale + causal mask
#pragma unroll
        for (int nt = 0; nt < 2; ++nt)
#pragma unroll
            for (int v = 0; v < 8; ++v) {
                float sv = s[nt][v] * scale;
                if (causal) {
                    int kg = k0 + nt * 16 + colc;
                    int qg = q0 + v + hi8;
                    if (kg > qg) sv = -1.0e30f;
                }
                s[nt][v] = sv;
            }

        // streaming softmax: row = v + hi8; its 16 columns live across a 16-lane group
        float alpha[8];
#pragma unroll
        for (int v = 0; v < 8; ++v) {
            float t = fmaxf(s[0][v], s[1][v]);
#pragma unroll
            for (int o = 1; o < 16; o <<= 1) t = fmaxf(t, __shfl_xor(t, o, 32));
            float mn = fmaxf(m[v], t);
            alpha[v] = __expf(m[v] - mn);
            m[v] = mn;
            float p0 = __expf(s[0][v] - mn);
            float p1 = __expf(s[1][v] - mn);
            s[0][v] = p0; s[1][v] = p1;
            float rs = p0 + p1;
#pragma unroll
            for (int o = 1; o < 16; o <<= 1) rs += __shfl_xor(rs, o, 32);
            l[v] = l[v] * alpha[v] + rs;
        }
#pragma unroll
        for (int n = 0; n < 4; ++n)
#pragma unroll
            for (int v = 0; v < 8; ++v) acc[n][v] *= alpha[v];

        // P (C-layout) -> LDS -> A-fragment layout
#pragma unroll
        for (int v = 0; v < 8; ++v) {
            Ps[v + hi8][colc]      = (_Float16)s[0][v];
            Ps[v + hi8][16 + colc] = (_Float16)s[1][v];
        }
        __syncthreads();

        v8h plo = *(const v8h*)&Ps[rq][klo];
        v8h phi = *(const v8h*)&Ps[rq][klo + 16];
        v16h aP = cat8(plo, phi);
#pragma unroll
        for (int n = 0; n < 4; ++n) {
            v8h vlo = *(const v8h*)&Vt[n * 16 + colc][klo];
            v8h vhi = *(const v8h*)&Vt[n * 16 + colc][klo + 16];
            v16h bV = cat8(vlo, vhi);
            acc[n] = __builtin_amdgcn_wmma_f32_16x16x32_f16(
                false, aP, false, bV, (short)0, acc[n], false, false);
        }
        __syncthreads();
    }

    // finalize and store into concat layout [row][h*64 + a]
#pragma unroll
    for (int v = 0; v < 8; ++v) {
        float inv = 1.0f / l[v];
#pragma unroll
        for (int n = 0; n < 4; ++n) {
            float val = acc[n][v] * inv;
            long row = (long)(b * Sc + q0 + v + hi8);
            int  col = h * Ac + n * 16 + colc;
            if (outH) outH[row * Ec + col] = (_Float16)val;
            if (outF) outF[row * Ec + col] = val;
        }
    }
}

// ---------------------------------------------------------------------------
// Fused residual add + LayerNorm; optional f16 shadow output.
// One 256-thread block per row of E=1024.
// ---------------------------------------------------------------------------
__global__ __launch_bounds__(256) void ln_add(
    const float* __restrict__ a, const float* __restrict__ bsrc,
    const float* __restrict__ g, const float* __restrict__ beta,
    float* __restrict__ outF, _Float16* __restrict__ outH) {
    const int row = blockIdx.x;
    const int t   = threadIdx.x;
    __shared__ float red[256];
    const float* pa = a + (long)row * Ec;
    const float* pb = bsrc + (long)row * Ec;
    float vals[4];
    float s = 0.0f;
#pragma unroll
    for (int i = 0; i < 4; ++i) { vals[i] = pa[t + i * 256] + pb[t + i * 256]; s += vals[i]; }
    red[t] = s; __syncthreads();
    for (int o = 128; o > 0; o >>= 1) { if (t < o) red[t] += red[t + o]; __syncthreads(); }
    float mu = red[0] * (1.0f / Ec);
    __syncthreads();
    float vs = 0.0f;
#pragma unroll
    for (int i = 0; i < 4; ++i) { float d = vals[i] - mu; vs += d * d; }
    red[t] = vs; __syncthreads();
    for (int o = 128; o > 0; o >>= 1) { if (t < o) red[t] += red[t + o]; __syncthreads(); }
    float rstd = rsqrtf(red[0] * (1.0f / Ec) + 1e-5f);
#pragma unroll
    for (int i = 0; i < 4; ++i) {
        int c = t + i * 256;
        float y = (vals[i] - mu) * rstd * g[c] + beta[c];
        outF[(long)row * Ec + c] = y;
        if (outH) outH[(long)row * Ec + c] = (_Float16)y;
    }
}

// ---------------------------------------------------------------------------
// Host launcher
// ---------------------------------------------------------------------------
extern "C" void kernel_launch(void* const* d_in, const int* in_sizes, int n_in,
                              void* d_out, int out_size, void* d_ws, size_t ws_size,
                              hipStream_t stream) {
    const float* emb   = (const float*)d_in[0];
    const float* pos   = (const float*)d_in[1];
    const float* kvk   = (const float*)d_in[2];
    const float* kvv   = (const float*)d_in[3];
    const float* Wq    = (const float*)d_in[4];
    const float* Wk    = (const float*)d_in[5];
    const float* Wv    = (const float*)d_in[6];
    const float* Wproj = (const float*)d_in[7];
    const float* ffW1  = (const float*)d_in[8];
    const float* ffb1  = (const float*)d_in[9];
    const float* ffW2  = (const float*)d_in[10];
    const float* ffb2  = (const float*)d_in[11];
    const float* ln_in_g  = (const float*)d_in[12];
    const float* ln_in_b  = (const float*)d_in[13];
    const float* ln_ed_g  = (const float*)d_in[14];
    const float* ln_ed_b  = (const float*)d_in[15];
    const float* ln_ff_g  = (const float*)d_in[16];
    const float* ln_ff_b  = (const float*)d_in[17];
    const float* ln_out_g = (const float*)d_in[18];
    const float* ln_out_b = (const float*)d_in[19];
    float* out = (float*)d_out;

    char* p = (char*)d_ws;
    auto alloc = [&](size_t bytes) -> void* {
        void* r = (void*)p;
        p += (bytes + 255) & ~(size_t)255;
        return r;
    };
    const size_t ME = (size_t)Mr * Ec;                 // 4M elements
    const size_t KV = (size_t)Hc * Bc * SKVc * Ac;     // 4M elements

    float*    xf     = (float*)alloc(ME * 4);
    _Float16* xh     = (_Float16*)alloc(ME * 2);
    _Float16* Wqh    = (_Float16*)alloc((size_t)Ec * Ec * 2);
    _Float16* Wkh    = (_Float16*)alloc((size_t)Ec * Ec * 2);
    _Float16* Wvh    = (_Float16*)alloc((size_t)Ec * Ec * 2);
    _Float16* Wph    = (_Float16*)alloc((size_t)Ec * Ec * 2);
    _Float16* W1h    = (_Float16*)alloc((size_t)Ec * FFc * 2);
    _Float16* W2h    = (_Float16*)alloc((size_t)FFc * Ec * 2);
    _Float16* kvkh   = (_Float16*)alloc(KV * 2);
    _Float16* kvvh   = (_Float16*)alloc(KV * 2);
    _Float16* Qh     = (_Float16*)alloc(ME * 2);
    _Float16* Kh     = (_Float16*)alloc(ME * 2);
    _Float16* Vh     = (_Float16*)alloc(ME * 2);
    _Float16* attnh  = (_Float16*)alloc(ME * 2);
    float*    projf  = (float*)alloc(ME * 4);
    float*    mhaf   = (float*)alloc(ME * 4);
    float*    xof    = (float*)alloc(ME * 4);
    float*    faf    = (float*)alloc(ME * 4);
    _Float16* fah    = (_Float16*)alloc(ME * 2);
    _Float16* hidh   = (_Float16*)alloc((size_t)Mr * FFc * 2);
    float*    ffof   = (float*)alloc(ME * 4);
    float*    decf   = (float*)alloc(ME * 4);

    const int T = 256;
    // x = emb + pos (f32 + f16 copies)
    add_cast_kernel<<<(int)((ME + T - 1) / T), T, 0, stream>>>(emb, pos, xf, xh, (int)ME);
    // weight repack / casts
    repack_qkv_kernel<<<(Ec * Ec + T - 1) / T, T, 0, stream>>>(Wq, Wqh, Ec * Ec);
    repack_qkv_kernel<<<(Ec * Ec + T - 1) / T, T, 0, stream>>>(Wk, Wkh, Ec * Ec);
    repack_qkv_kernel<<<(Ec * Ec + T - 1) / T, T, 0, stream>>>(Wv, Wvh, Ec * Ec);
    cast_kernel<<<(Ec * Ec + T - 1) / T, T, 0, stream>>>(Wproj, Wph, Ec * Ec);
    cast_kernel<<<(int)((Ec * (size_t)FFc + T - 1) / T), T, 0, stream>>>(ffW1, W1h, Ec * FFc);
    cast_kernel<<<(int)((Ec * (size_t)FFc + T - 1) / T), T, 0, stream>>>(ffW2, W2h, FFc * Ec);
    cast_kernel<<<(int)((KV + T - 1) / T), T, 0, stream>>>(kvk, kvkh, (int)KV);
    cast_kernel<<<(int)((KV + T - 1) / T), T, 0, stream>>>(kvv, kvvh, (int)KV);

    // Q, K, V projections
    dim3 g1(Ec / 128, Mr / 128);
    gemm_wmma<1, 0, 0><<<g1, 256, 0, stream>>>(xh, Wqh, nullptr, Qh, Mr, Ec, Ec);
    gemm_wmma<1, 0, 0><<<g1, 256, 0, stream>>>(xh, Wkh, nullptr, Kh, Mr, Ec, Ec);
    gemm_wmma<1, 0, 0><<<g1, 256, 0, stream>>>(xh, Wvh, nullptr, Vh, Mr, Ec, Ec);

    // causal self-attention (scale 1/sqrt(E) = 1/32)
    dim3 ga(Sc / 16, Hc * Bc);
    flash_attn<<<ga, 32, 0, stream>>>(Qh, Kh, Vh, attnh, nullptr, Sc, 1, 1, 0.03125f);

    // projection + residual LN
    gemm_wmma<0, 0, 0><<<g1, 256, 0, stream>>>(attnh, Wph, nullptr, projf, Mr, Ec, Ec);
    ln_add<<<Mr, 256, 0, stream>>>(xf, projf, ln_in_g, ln_in_b, mhaf, nullptr);

    // cross-attention (no mask, scale 1/sqrt(SKV) = 1/32)
    flash_attn<<<ga, 32, 0, stream>>>(Qh, kvkh, kvvh, nullptr, xof, SKVc, 0, 0, 0.03125f);
    ln_add<<<Mr, 256, 0, stream>>>(xof, mhaf, ln_ed_g, ln_ed_b, faf, fah);

    // feed-forward: hidden = fa @ W1 + b1 ; ff_out = relu(hidden @ W2 + b2)
    dim3 g2(FFc / 128, Mr / 128);
    gemm_wmma<1, 0, 1><<<g2, 256, 0, stream>>>(fah, W1h, ffb1, hidh, Mr, FFc, Ec);
    gemm_wmma<0, 1, 1><<<g1, 256, 0, stream>>>(hidh, W2h, ffb2, ffof, Mr, Ec, FFc);

    // decoded = LN(fa + ff_out) ; out = LN(fa + decoded)
    ln_add<<<Mr, 256, 0, stream>>>(faf, ffof, ln_ff_g, ln_ff_b, decf, nullptr);
    ln_add<<<Mr, 256, 0, stream>>>(faf, decf, ln_out_g, ln_out_b, out, nullptr);

    (void)in_sizes; (void)n_in; (void)out_size; (void)ws_size;
}
